// VisDescMatcher_35373350650684
// MI455X (gfx1250) — matrix operand order repaired and checked by reading, not compiled, not run on hardware
//
#include <hip/hip_runtime.h>

// Problem constants (match reference)
#define B_   16
#define N2_  1024
#define N3_  2048
#define D_   128
#define ON3  2049   // N3 + 1 (dustbin row)
#define ON2  1025   // N2 + 1 (dustbin col)

#define TM   128    // M tile per workgroup
#define TN   128    // N tile per workgroup
#define LDK  264    // 256 virtual-K (hi|lo) + 8 bf16 pad (16B) per LDS row

typedef __attribute__((ext_vector_type(16))) __bf16 v16bf;
typedef __attribute__((ext_vector_type(8)))  float  v8f;

union FragBF { uint4 q[2]; v16bf v; };

static __device__ __forceinline__ unsigned short f2bf_rne(float f) {
    unsigned u = __builtin_bit_cast(unsigned, f);
    u += 0x7FFFu + ((u >> 16) & 1u);          // round-to-nearest-even
    return (unsigned short)(u >> 16);
}
static __device__ __forceinline__ float bf2f(unsigned short h) {
    unsigned u = ((unsigned)h) << 16;
    return __builtin_bit_cast(float, u);
}

// ---------------------------------------------------------------------------
// Kernel 1: per-row inverse L2 norm.  One wave32 per row of 128 floats.
// ---------------------------------------------------------------------------
__global__ void vdm_rownorm(const float* __restrict__ x,
                            float* __restrict__ inv, int nrows) {
    int wave = (blockIdx.x * blockDim.x + threadIdx.x) >> 5;
    int lane = threadIdx.x & 31;
    if (wave >= nrows) return;
    float4 v = ((const float4*)(x + (size_t)wave * D_))[lane];
    float s = v.x * v.x + v.y * v.y + v.z * v.z + v.w * v.w;
#pragma unroll
    for (int off = 16; off > 0; off >>= 1) s += __shfl_xor(s, off, 32);
    if (lane == 0) inv[wave] = rsqrtf(s);
}

// ---------------------------------------------------------------------------
// Kernel 2: batched GEMM sim[b] = d3n[b] (2048x128) * d2n[b]^T (128x1024)
// via bf16 hi/lo split (virtual K = 256) and v_wmma_f32_16x16x32_bf16.
// Workgroup: 256 threads = 8 waves, 128x128 output tile, LDS-staged operands.
// ---------------------------------------------------------------------------
__global__ __launch_bounds__(256)
void vdm_gemm(const float* __restrict__ d3, const float* __restrict__ d2,
              const float* __restrict__ inv3, const float* __restrict__ inv2,
              float* __restrict__ out) {
    __shared__ unsigned short sA[TM * LDK];   // d3 tile, bf16 hi|lo
    __shared__ unsigned short sB[TN * LDK];   // d2 tile, bf16 hi|lo

    const int b  = blockIdx.z;
    const int m0 = blockIdx.y * TM;
    const int n0 = blockIdx.x * TN;

    const float* Ag = d3   + ((size_t)b * N3_ + m0) * D_;
    const float* Bg = d2   + ((size_t)b * N2_ + n0) * D_;
    const float* ia = inv3 + (size_t)b * N3_ + m0;
    const float* ib = inv2 + (size_t)b * N2_ + n0;

    // ---- stage: load f32, scale by inv-norm, split into bf16 hi (cols 0..127)
    //      and bf16 lo (cols 128..255), store to LDS ----
#pragma unroll 4
    for (int i = threadIdx.x; i < TM * (D_ / 4); i += 256) {
        int row = i >> 5, c4 = i & 31;
        float4 v = ((const float4*)(Ag + row * D_))[c4];
        float s = ia[row];
        float x0 = v.x * s, x1 = v.y * s, x2 = v.z * s, x3 = v.w * s;
        unsigned short h0 = f2bf_rne(x0), h1 = f2bf_rne(x1),
                       h2 = f2bf_rne(x2), h3 = f2bf_rne(x3);
        unsigned short l0 = f2bf_rne(x0 - bf2f(h0)), l1 = f2bf_rne(x1 - bf2f(h1)),
                       l2 = f2bf_rne(x2 - bf2f(h2)), l3 = f2bf_rne(x3 - bf2f(h3));
        uint2 hp, lp;
        hp.x = (unsigned)h0 | ((unsigned)h1 << 16);
        hp.y = (unsigned)h2 | ((unsigned)h3 << 16);
        lp.x = (unsigned)l0 | ((unsigned)l1 << 16);
        lp.y = (unsigned)l2 | ((unsigned)l3 << 16);
        *(uint2*)&sA[row * LDK + c4 * 4]        = hp;
        *(uint2*)&sA[row * LDK + 128 + c4 * 4]  = lp;
    }
#pragma unroll 4
    for (int i = threadIdx.x; i < TN * (D_ / 4); i += 256) {
        int row = i >> 5, c4 = i & 31;
        float4 v = ((const float4*)(Bg + row * D_))[c4];
        float s = ib[row];
        float x0 = v.x * s, x1 = v.y * s, x2 = v.z * s, x3 = v.w * s;
        unsigned short h0 = f2bf_rne(x0), h1 = f2bf_rne(x1),
                       h2 = f2bf_rne(x2), h3 = f2bf_rne(x3);
        unsigned short l0 = f2bf_rne(x0 - bf2f(h0)), l1 = f2bf_rne(x1 - bf2f(h1)),
                       l2 = f2bf_rne(x2 - bf2f(h2)), l3 = f2bf_rne(x3 - bf2f(h3));
        uint2 hp, lp;
        hp.x = (unsigned)h0 | ((unsigned)h1 << 16);
        hp.y = (unsigned)h2 | ((unsigned)h3 << 16);
        lp.x = (unsigned)l0 | ((unsigned)l1 << 16);
        lp.y = (unsigned)l2 | ((unsigned)l3 << 16);
        *(uint2*)&sB[row * LDK + c4 * 4]        = hp;
        *(uint2*)&sB[row * LDK + 128 + c4 * 4]  = lp;
    }
    __syncthreads();

    // ---- compute: wave w owns 32(M) x 64(N) sub-tile ----
    const int lane = threadIdx.x & 31;
    const int w    = threadIdx.x >> 5;
    const int mw   = (w & 3) * 32;
    const int nw   = (w >> 2) * 64;
    const int l16  = lane & 15;
    const int lh   = lane >> 4;

    v8f acc[2][4];
#pragma unroll
    for (int mi = 0; mi < 2; ++mi)
#pragma unroll
        for (int ni = 0; ni < 4; ++ni) acc[mi][ni] = 0.0f;

#pragma unroll
    for (int ks = 0; ks < 8; ++ks) {   // virtual K = 256, step 32
        const int kb = ks * 32;
        FragBF a[2], bfr[4];
#pragma unroll
        for (int mi = 0; mi < 2; ++mi) {
            // A layout: M = lane%16; K chunks at kb+8*(lane/16) and +16
            const unsigned short* p =
                &sA[(mw + mi * 16 + l16) * LDK + kb + lh * 8];
            a[mi].q[0] = *(const uint4*)p;
            a[mi].q[1] = *(const uint4*)(p + 16);
        }
#pragma unroll
        for (int ni = 0; ni < 4; ++ni) {
            // B layout: N = lane%16; K chunks at kb+8*(lane/16) and +16
            const unsigned short* p =
                &sB[(nw + ni * 16 + l16) * LDK + kb + lh * 8];
            bfr[ni].q[0] = *(const uint4*)p;
            bfr[ni].q[1] = *(const uint4*)(p + 16);
        }
#pragma unroll
        for (int mi = 0; mi < 2; ++mi)
#pragma unroll
            for (int ni = 0; ni < 4; ++ni)
                acc[mi][ni] = __builtin_amdgcn_wmma_f32_16x16x32_bf16(
                    false, a[mi].v, false, bfr[ni].v,
                    (short)0, acc[mi][ni], false, false);
    }

    // ---- write out: C/D layout N = lane%16, M = r + 8*(lane/16) ----
    float* ob = out + (size_t)b * ON3 * ON2;
#pragma unroll
    for (int mi = 0; mi < 2; ++mi)
#pragma unroll
        for (int ni = 0; ni < 4; ++ni) {
            int col  = n0 + nw + ni * 16 + l16;
            int row0 = m0 + mw + mi * 16 + lh * 8;
#pragma unroll
            for (int r = 0; r < 8; ++r)
                ob[(size_t)(row0 + r) * ON2 + col] = acc[mi][ni][r];
        }
}

// ---------------------------------------------------------------------------
// Kernel 3: zero the dustbin row/column (harness poisons d_out each run).
// ---------------------------------------------------------------------------
__global__ void vdm_pad_zero(float* __restrict__ out) {
    int b = blockIdx.y;
    int i = blockIdx.x * 256 + threadIdx.x;
    size_t base = (size_t)b * ON3 * ON2;
    if (i < ON3) {
        out[base + (size_t)i * ON2 + (ON2 - 1)] = 0.0f;          // last column
    } else if (i < ON3 + N2_) {
        out[base + (size_t)(ON3 - 1) * ON2 + (i - ON3)] = 0.0f;  // last row
    }
}

// ---------------------------------------------------------------------------
extern "C" void kernel_launch(void* const* d_in, const int* in_sizes, int n_in,
                              void* d_out, int out_size, void* d_ws, size_t ws_size,
                              hipStream_t stream) {
    const float* desc2d = (const float*)d_in[0];
    const float* desc3d = (const float*)d_in[2];
    float* out  = (float*)d_out;
    float* inv3 = (float*)d_ws;                     // B*N3 floats
    float* inv2 = inv3 + (size_t)B_ * N3_;          // B*N2 floats

    // inverse norms: one wave per row, 8 waves (256 threads) per block
    vdm_rownorm<<<(B_ * N3_ + 7) / 8, 256, 0, stream>>>(desc3d, inv3, B_ * N3_);
    vdm_rownorm<<<(B_ * N2_ + 7) / 8, 256, 0, stream>>>(desc2d, inv2, B_ * N2_);

    dim3 grid(N2_ / TN, N3_ / TM, B_);              // 8 x 16 x 16 workgroups
    vdm_gemm<<<grid, 256, 0, stream>>>(desc3d, desc2d, inv3, inv2, out);

    dim3 pgrid((ON3 + N2_ + 255) / 256, B_);
    vdm_pad_zero<<<pgrid, 256, 0, stream>>>(out);
}
